// Net_59407987638976
// MI455X (gfx1250) — compile-verified, matrix operand order checked by reading, not run on hardware
//
#include <hip/hip_runtime.h>
#include <hip/hip_bf16.h>
#include <stdint.h>

// ---------------------------------------------------------------------------
// 2-layer LSTM (B=1024, T=512, H=256) for gfx1250.
//  * prep_day: fold day-embedding + biases of layer-0 input proj into 7x1024
//    table (day in {0..6}) -> removes the FEAT=65 GEMM entirely.
//  * pack_b:   repack Whh0/Wih1/Whh1 (1024x256 f32) into bf16 WMMA B-fragment
//    order: 1KB contiguous per (ntile,ktile) fragment -> 2x global_load_b128
//    per lane in the hot loop.
//  * lstm2:    persistent kernel, 32 WGs x 8 waves; each WG owns 32 batch rows.
//    h0/h1 double-buffered bf16 in LDS (64KB); c0/c1 in VGPRs in WMMA C-layout.
//    Each B-fragment is reused for both 16-row batch tiles (acc[2][4]) so each
//    wave streams its weight columns exactly once per timestep (all L2-hits).
//    K loops unrolled only 2x to keep VGPR liveness ~200 (no scratch spills).
// ---------------------------------------------------------------------------

typedef __attribute__((ext_vector_type(16))) __bf16 v16bf;
typedef __attribute__((ext_vector_type(8)))  float  v8f;

#define HH   256   // hidden
#define TT   512   // timesteps
#define BT   32    // batch rows per workgroup

union AF { v16bf v; uint4 q[2]; };

// A fragment (16x32 bf16) from row-major LDS h[32][256].
// lanes 0-15: M=lane, K = kt*32 + {0..7,16..23}; lanes 16-31: same M, K+8.
__device__ __forceinline__ v16bf loadA_lds(const __bf16* h, int rt, int kt, int lane) {
  const int row = rt * 16 + (lane & 15);
  const int col = kt * 32 + ((lane & 16) ? 8 : 0);
  const __bf16* p = h + row * HH + col;
  AF a;
  a.q[0] = *(const uint4*)(p);
  a.q[1] = *(const uint4*)(p + 16);
  return a.v;
}

// B fragment (32x16 bf16) from pre-packed weights: contiguous 1KB per fragment.
__device__ __forceinline__ v16bf loadB_pk(const __bf16* __restrict__ Wp,
                                          int gnt, int kt, int lane) {
  const __bf16* p = Wp + (((size_t)gnt * 8 + kt) * 32 + lane) * 16;
  AF b;
  b.q[0] = *(const uint4*)(p);
  b.q[1] = *(const uint4*)(p + 8);
  return b.v;
}

__device__ __forceinline__ float sigm(float x) {
  float e = __expf(-x);
  return __builtin_amdgcn_rcpf(1.0f + e);
}
__device__ __forceinline__ float tanhx(float x) {
  x = fminf(fmaxf(x, -12.0f), 12.0f);
  float e = __expf(2.0f * x);
  return (e - 1.0f) * __builtin_amdgcn_rcpf(e + 1.0f);
}

// ---------------------------------------------------------------------------
// Prep kernel 1: dayTab[d][g] = sum_k Wih0[g][1+k]*emb[d][k] + bih0[g]+bhh0[g]
// Also wcol0[g] = Wih0[g][0], bias1[g] = bih1[g]+bhh1[g].
// ---------------------------------------------------------------------------
__global__ void prep_day(const float* __restrict__ Wih0, const float* __restrict__ emb,
                         const float* __restrict__ bih0, const float* __restrict__ bhh0,
                         const float* __restrict__ bih1, const float* __restrict__ bhh1,
                         float* __restrict__ dayTab, float* __restrict__ wcol0,
                         float* __restrict__ bias1) {
  int id = blockIdx.x * 256 + threadIdx.x;
  if (id < 7 * 1024) {
    int d = id >> 10, g = id & 1023;
    float acc = bih0[g] + bhh0[g];
    #pragma unroll 8
    for (int k = 0; k < 64; ++k) acc += Wih0[g * 65 + 1 + k] * emb[d * 64 + k];
    dayTab[id] = acc;
  }
  if (id < 1024) {
    wcol0[id] = Wih0[id * 65];
    bias1[id] = bih1[id] + bhh1[id];
  }
}

// ---------------------------------------------------------------------------
// Prep kernel 2: pack W (1024x256 f32, row-major, B = W^T) into bf16 fragments.
// dst element index == thread id: id = ((ntile*8 + kt)*32 + lane)*16 + e
// ---------------------------------------------------------------------------
__global__ void pack_b(const float* __restrict__ W, __bf16* __restrict__ out) {
  int id = blockIdx.x * 256 + threadIdx.x;          // 262144 total
  int e     = id & 15;
  int lane  = (id >> 4) & 31;
  int kt    = (id >> 9) & 7;
  int ntile = id >> 12;                              // 0..63
  int n   = ntile * 16 + (lane & 15);
  int kin = (e < 8 ? e : e + 8) + ((lane & 16) ? 8 : 0);
  int K   = kt * 32 + kin;
  out[id] = (__bf16)W[n * 256 + K];
}

// ---------------------------------------------------------------------------
// Main persistent LSTM kernel: grid = 32 blocks x 256 threads.
// ---------------------------------------------------------------------------
__global__ __launch_bounds__(256, 1)
void lstm2_kernel(const float* __restrict__ x,        // (B,T,2)
                  const float* __restrict__ dayTab,   // (7,1024), bias0 folded
                  const float* __restrict__ wcol0,    // (1024)
                  const float* __restrict__ bias1,    // (1024)
                  const __bf16* __restrict__ Whh0p,
                  const __bf16* __restrict__ Wih1p,
                  const __bf16* __restrict__ Whh1p,
                  const float* __restrict__ Wlin,     // (14,256)
                  const float* __restrict__ blin,     // (14)
                  float* __restrict__ out)            // (B,14)
{
  __shared__ __bf16 h0s[2][BT][HH];   // 32 KB (double-buffered)
  __shared__ __bf16 h1s[2][BT][HH];   // 32 KB

  const int tid  = threadIdx.x;
  const int w    = tid >> 5;          // wave 0..7 -> owns h cols [w*32, w*32+32)
  const int lane = tid & 31;
  const int ln   = lane & 15;
  const int rb   = (lane & 16) ? 8 : 0;   // C-layout row offset for hi lanes
  const int b0   = blockIdx.x * BT;

  {
    __bf16* z0 = &h0s[0][0][0];
    __bf16* z1 = &h1s[0][0][0];
    for (int i = tid; i < 2 * BT * HH; i += 256) { z0[i] = (__bf16)0.0f; z1[i] = (__bf16)0.0f; }
  }

  // Per-lane column constants for the wave's gate columns.
  float wc0[4][2], b1c[4][2];
  #pragma unroll
  for (int c = 0; c < 4; ++c)
    #pragma unroll
    for (int nt = 0; nt < 2; ++nt) {
      int col = c * 256 + w * 32 + nt * 16 + ln;
      wc0[c][nt] = wcol0[col];
      b1c[c][nt] = bias1[col];
    }

  // Cell state in WMMA C-layout: [rowtile][ntile] -> 8 f32 per lane.
  v8f c0st[2][2], c1st[2][2];
  #pragma unroll
  for (int rt = 0; rt < 2; ++rt)
    #pragma unroll
    for (int nt = 0; nt < 2; ++nt)
      #pragma unroll
      for (int r = 0; r < 8; ++r) { c0st[rt][nt][r] = 0.0f; c1st[rt][nt][r] = 0.0f; }

  __syncthreads();

  int cur = 0;
  for (int t = 0; t < TT; ++t) {
    const int nxt = cur ^ 1;

    // Each lane loads (val, day) for batch row b0+lane of this timestep.
    float2 xv2 = *(const float2*)(x + ((size_t)(b0 + lane) * TT + t) * 2);
    float vloc = xv2.x;
    int   dloc = ((int)xv2.y) * 1024;

    // ---------------- layer 0: gates = init + h0 @ Whh0^T ----------------
    #pragma unroll
    for (int nt = 0; nt < 2; ++nt) {
      v8f acc[2][4];
      #pragma unroll
      for (int rt = 0; rt < 2; ++rt)
        #pragma unroll
        for (int r = 0; r < 8; ++r) {
          int row = rt * 16 + r + rb;
          float xv = __shfl(vloc, row, 32);
          int   xo = __shfl(dloc, row, 32);
          #pragma unroll
          for (int c = 0; c < 4; ++c) {
            int col = c * 256 + w * 32 + nt * 16 + ln;
            acc[rt][c][r] = xv * wc0[c][nt] + dayTab[xo + col];
          }
        }
      #pragma unroll 2
      for (int kt = 0; kt < 8; ++kt) {
        v16bf a0 = loadA_lds(&h0s[cur][0][0], 0, kt, lane);
        v16bf a1 = loadA_lds(&h0s[cur][0][0], 1, kt, lane);
        #pragma unroll
        for (int c = 0; c < 4; ++c) {
          v16bf bf = loadB_pk(Whh0p, c * 16 + w * 2 + nt, kt, lane);
          acc[0][c] = __builtin_amdgcn_wmma_f32_16x16x32_bf16(
              false, a0, false, bf, (short)0, acc[0][c], false, false);
          acc[1][c] = __builtin_amdgcn_wmma_f32_16x16x32_bf16(
              false, a1, false, bf, (short)0, acc[1][c], false, false);
        }
      }
      #pragma unroll
      for (int rt = 0; rt < 2; ++rt) {
        v8f cst = c0st[rt][nt];
        #pragma unroll
        for (int r = 0; r < 8; ++r) {
          float ig = sigm(acc[rt][0][r]);
          float fg = sigm(acc[rt][1][r]);
          float gg = tanhx(acc[rt][2][r]);
          float og = sigm(acc[rt][3][r]);
          float cv = fg * cst[r] + ig * gg;
          cst[r] = cv;
          float hv = og * tanhx(cv);
          h0s[nxt][rt * 16 + r + rb][w * 32 + nt * 16 + ln] = (__bf16)hv;
        }
        c0st[rt][nt] = cst;
      }
    }
    __syncthreads();   // h0[nxt] complete before layer 1 consumes it

    // -------- layer 1: gates = b1 + h0_new @ Wih1^T + h1 @ Whh1^T --------
    #pragma unroll
    for (int nt = 0; nt < 2; ++nt) {
      v8f acc[2][4];
      #pragma unroll
      for (int rt = 0; rt < 2; ++rt)
        #pragma unroll
        for (int c = 0; c < 4; ++c)
          #pragma unroll
          for (int r = 0; r < 8; ++r) acc[rt][c][r] = b1c[c][nt];
      #pragma unroll 2
      for (int kt = 0; kt < 8; ++kt) {
        v16bf a0 = loadA_lds(&h0s[nxt][0][0], 0, kt, lane);
        v16bf a1 = loadA_lds(&h0s[nxt][0][0], 1, kt, lane);
        #pragma unroll
        for (int c = 0; c < 4; ++c) {
          v16bf bf = loadB_pk(Wih1p, c * 16 + w * 2 + nt, kt, lane);
          acc[0][c] = __builtin_amdgcn_wmma_f32_16x16x32_bf16(
              false, a0, false, bf, (short)0, acc[0][c], false, false);
          acc[1][c] = __builtin_amdgcn_wmma_f32_16x16x32_bf16(
              false, a1, false, bf, (short)0, acc[1][c], false, false);
        }
      }
      #pragma unroll 2
      for (int kt = 0; kt < 8; ++kt) {
        v16bf a0 = loadA_lds(&h1s[cur][0][0], 0, kt, lane);
        v16bf a1 = loadA_lds(&h1s[cur][0][0], 1, kt, lane);
        #pragma unroll
        for (int c = 0; c < 4; ++c) {
          v16bf bf = loadB_pk(Whh1p, c * 16 + w * 2 + nt, kt, lane);
          acc[0][c] = __builtin_amdgcn_wmma_f32_16x16x32_bf16(
              false, a0, false, bf, (short)0, acc[0][c], false, false);
          acc[1][c] = __builtin_amdgcn_wmma_f32_16x16x32_bf16(
              false, a1, false, bf, (short)0, acc[1][c], false, false);
        }
      }
      #pragma unroll
      for (int rt = 0; rt < 2; ++rt) {
        v8f cst = c1st[rt][nt];
        #pragma unroll
        for (int r = 0; r < 8; ++r) {
          float ig = sigm(acc[rt][0][r]);
          float fg = sigm(acc[rt][1][r]);
          float gg = tanhx(acc[rt][2][r]);
          float og = sigm(acc[rt][3][r]);
          float cv = fg * cst[r] + ig * gg;
          cst[r] = cv;
          float hv = og * tanhx(cv);
          h1s[nxt][rt * 16 + r + rb][w * 32 + nt * 16 + ln] = (__bf16)hv;
        }
        c1st[rt][nt] = cst;
      }
    }
    __syncthreads();   // h1[nxt]/h0[nxt] reads done before next step overwrites
    cur = nxt;
  }

  // ---------------- head: out = h1_last @ Wlin^T + blin ----------------
  for (int idx = tid; idx < BT * 14; idx += 256) {
    int row = idx / 14, p = idx % 14;
    float acc = blin[p];
    #pragma unroll 8
    for (int k = 0; k < HH; ++k)
      acc += (float)h1s[cur][row][k] * Wlin[p * HH + k];
    out[(size_t)(b0 + row) * 14 + p] = acc;
  }
}

// ---------------------------------------------------------------------------
extern "C" void kernel_launch(void* const* d_in, const int* in_sizes, int n_in,
                              void* d_out, int out_size, void* d_ws, size_t ws_size,
                              hipStream_t stream) {
  const float* x    = (const float*)d_in[0];
  const float* emb  = (const float*)d_in[1];
  const float* Wih0 = (const float*)d_in[2];
  const float* Whh0 = (const float*)d_in[3];
  const float* bih0 = (const float*)d_in[4];
  const float* bhh0 = (const float*)d_in[5];
  const float* Wih1 = (const float*)d_in[6];
  const float* Whh1 = (const float*)d_in[7];
  const float* bih1 = (const float*)d_in[8];
  const float* bhh1 = (const float*)d_in[9];
  const float* Wlin = (const float*)d_in[10];
  const float* blin = (const float*)d_in[11];
  float* out = (float*)d_out;

  char* ws = (char*)d_ws;
  float*  dayTab = (float*)(ws);                       // 28672 B
  float*  wcol0  = (float*)(ws + 28672);               // 4096 B
  float*  bias1  = (float*)(ws + 32768);               // 4096 B
  __bf16* Whh0p  = (__bf16*)(ws + 36864);              // 512 KB
  __bf16* Wih1p  = (__bf16*)(ws + 36864 + 524288);     // 512 KB
  __bf16* Whh1p  = (__bf16*)(ws + 36864 + 2 * 524288); // 512 KB

  prep_day<<<28, 256, 0, stream>>>(Wih0, emb, bih0, bhh0, bih1, bhh1,
                                   dayTab, wcol0, bias1);
  pack_b<<<1024, 256, 0, stream>>>(Whh0, Whh0p);
  pack_b<<<1024, 256, 0, stream>>>(Wih1, Wih1p);
  pack_b<<<1024, 256, 0, stream>>>(Whh1, Whh1p);
  lstm2_kernel<<<32, 256, 0, stream>>>(x, dayTab, wcol0, bias1,
                                       Whh0p, Wih1p, Whh1p, Wlin, blin, out);
}
